// PatchFeatureExtractor_16638703305432
// MI455X (gfx1250) — matrix operand order repaired and checked by reading, not compiled, not run on hardware
//
#include <hip/hip_runtime.h>

// ---------------------------------------------------------------------------
// DGCNN edge-conv block for MI455X (gfx1250, wave32, WMMA).
// Pipeline (all on `stream`, deterministic, no FP atomics):
//   1. knn            : top-20 neighbor indices per point (8-way split scan)
//   2. feat_conv1     : build (xj-xi, xi) features, 6->64 conv, store h1 f16
//   3. stats/finalize : BN1 stats (block partials + sequential reduce)
//   4. layer2 (WMMA)  : BN1+ReLU on the fly, 64->128 f16 WMMA, store h2 f16
//   5. stats/finalize : BN2
//   6. layer3 (WMMA)  : BN2+ReLU, 128->128 f16 WMMA, store h3 f16
//   7. stats/finalize : BN3
//   8. maxpool        : BN3 + max over k, transposed coalesced output
// Roofline: ~500MB total HBM traffic -> ~20us at 23.3 TB/s; GEMM FLOPs are
// negligible vs WMMA throughput, so everything is bandwidth/selection bound.
// ---------------------------------------------------------------------------

#define BB    4
#define NPTS  4096
#define KNN   20
#define PPOS  (BB * NPTS * KNN)   // 327680 positions
#define C1    64
#define C2    128
#define C3    128
#define EPSBN 1e-5f

typedef _Float16 v16h __attribute__((ext_vector_type(16)));
typedef float    v8f  __attribute__((ext_vector_type(8)));
typedef _Float16 h8   __attribute__((ext_vector_type(8)));

// ---------------------------------------------------------------------------
// 1. KNN: 8 threads per query, each scans 512 candidates with a register
//    insertion-sorted top-20, then one thread merges the 8 sorted lists.
// ---------------------------------------------------------------------------
#define PARTS 8
#define QPB   32  // queries per block (256 threads)

__global__ __launch_bounds__(256) void knn_kernel(const float* __restrict__ x,
                                                  int* __restrict__ idx) {
  int b  = blockIdx.x / (NPTS / QPB);
  int n0 = (blockIdx.x % (NPTS / QPB)) * QPB;
  int tid  = threadIdx.x;
  int q    = tid / PARTS;
  int part = tid % PARTS;
  int n = n0 + q;
  const float* xb = x + (long)b * NPTS * 3;
  float xi0 = xb[n * 3 + 0], xi1 = xb[n * 3 + 1], xi2 = xb[n * 3 + 2];
  float xxi = xi0 * xi0 + xi1 * xi1 + xi2 * xi2;

  float vals[KNN];
  int   inds[KNN];
#pragma unroll
  for (int s = 0; s < KNN; s++) { vals[s] = -__builtin_inff(); inds[s] = -1; }

  int j0 = part * (NPTS / PARTS);
  for (int j = j0; j < j0 + NPTS / PARTS; j++) {
    float xj0 = xb[j * 3 + 0], xj1 = xb[j * 3 + 1], xj2 = xb[j * 3 + 2];
    float dot = xi0 * xj0 + xi1 * xj1 + xi2 * xj2;
    float xxj = xj0 * xj0 + xj1 * xj1 + xj2 * xj2;
    float v = 2.0f * dot - xxi - xxj;  // pd = -xx_i - inner - xx_j (<= 0)
    if (v > vals[KNN - 1]) {
      float cv = v; int ci = j;
#pragma unroll
      for (int s = 0; s < KNN; s++) {
        if (cv > vals[s]) {
          float tv = vals[s]; vals[s] = cv; cv = tv;
          int   ti = inds[s]; inds[s] = ci; ci = ti;
        }
      }
    }
  }

  __shared__ float smv[QPB][PARTS][KNN];   // 20 KB
  __shared__ int   smi[QPB][PARTS][KNN];   // 20 KB
#pragma unroll
  for (int s = 0; s < KNN; s++) { smv[q][part][s] = vals[s]; smi[q][part][s] = inds[s]; }
  __syncthreads();

  if (part == 0) {
    int head[PARTS];
    for (int t = 0; t < PARTS; t++) head[t] = 0;
    int* op = idx + ((long)b * NPTS + n) * KNN;
    for (int kk = 0; kk < KNN; kk++) {
      float best = -__builtin_inff(); int bt = 0;
      for (int t = 0; t < PARTS; t++) {
        int h = head[t];
        if (h < KNN) {
          float v = smv[q][t][h];
          if (v > best) { best = v; bt = t; }
        }
      }
      op[kk] = smi[q][bt][head[bt]];
      head[bt]++;
    }
  }
}

// ---------------------------------------------------------------------------
// 2. Feature build + conv1 (6 -> 64) per position, store h1 as f16 [P][64].
// ---------------------------------------------------------------------------
__global__ __launch_bounds__(256) void feat_conv1_kernel(
    const float* __restrict__ x, const int* __restrict__ idx,
    const float* __restrict__ W1, const float* __restrict__ b1,
    _Float16* __restrict__ h1) {
  __shared__ float sW[C1][6];
  __shared__ float sb[C1];
  int tid = threadIdx.x;
  for (int i = tid; i < C1 * 6; i += 256) sW[i / 6][i % 6] = W1[i];
  if (tid < C1) sb[tid] = b1[tid];
  __syncthreads();

  long p = (long)blockIdx.x * 256 + tid;          // grid covers P exactly
  int b   = (int)(p / (NPTS * KNN));
  int rem = (int)(p % (NPTS * KNN));
  int n   = rem / KNN;
  const float* xb = x + (long)b * NPTS * 3;
  int j = idx[p];
  float xi0 = xb[n * 3 + 0], xi1 = xb[n * 3 + 1], xi2 = xb[n * 3 + 2];
  float xj0 = xb[j * 3 + 0], xj1 = xb[j * 3 + 1], xj2 = xb[j * 3 + 2];
  float f0 = xj0 - xi0, f1 = xj1 - xi1, f2 = xj2 - xi2;
  float f3 = xi0, f4 = xi1, f5 = xi2;

  _Float16 outv[C1];
#pragma unroll
  for (int o = 0; o < C1; o++) {
    float acc = sb[o];
    acc += sW[o][0] * f0; acc += sW[o][1] * f1; acc += sW[o][2] * f2;
    acc += sW[o][3] * f3; acc += sW[o][4] * f4; acc += sW[o][5] * f5;
    outv[o] = (_Float16)acc;
  }
  h8* dst = (h8*)(h1 + p * C1);
#pragma unroll
  for (int v = 0; v < C1 / 8; v++) {
    h8 t;
#pragma unroll
    for (int e = 0; e < 8; e++) t[e] = outv[v * 8 + e];
    dst[v] = t;
  }
}

// ---------------------------------------------------------------------------
// 3. BN stats: deterministic block partial sums (sum, sumsq) per channel.
// ---------------------------------------------------------------------------
#define STATBLKS 512

template <int C>
__global__ __launch_bounds__(256) void stats_partial_kernel(
    const _Float16* __restrict__ h, float* __restrict__ partial) {
  const int G = 256 / C;
  int tid = threadIdx.x;
  int g = tid / C, c = tid % C;
  long start  = (long)blockIdx.x * G + g;
  long stride = (long)STATBLKS * G;
  float s = 0.f, ss = 0.f;
  for (long p = start; p < PPOS; p += stride) {
    float v = (float)h[p * C + c];
    s += v; ss += v * v;
  }
  __shared__ float rs[256], rss[256];
  rs[tid] = s; rss[tid] = ss;
  __syncthreads();
  if (tid < C) {
    float a = 0.f, b2 = 0.f;
    for (int gg = 0; gg < G; gg++) { a += rs[gg * C + tid]; b2 += rss[gg * C + tid]; }
    partial[(long)blockIdx.x * 2 * C + tid]     = a;
    partial[(long)blockIdx.x * 2 * C + C + tid] = b2;
  }
}

__global__ void stats_final_kernel(const float* __restrict__ partial, int C,
                                   const float* __restrict__ gamma,
                                   const float* __restrict__ beta,
                                   float* __restrict__ a_out,
                                   float* __restrict__ c_out) {
  int c = threadIdx.x;
  if (c >= C) return;
  float s = 0.f, ss = 0.f;
  for (int i = 0; i < STATBLKS; i++) {
    s  += partial[(long)i * 2 * C + c];
    ss += partial[(long)i * 2 * C + C + c];
  }
  const float invP = 1.0f / (float)PPOS;
  float mean = s * invP;
  float var  = ss * invP - mean * mean;
  float a = gamma[c] * rsqrtf(var + EPSBN);
  a_out[c] = a;
  c_out[c] = beta[c] - mean * a;
}

// ---------------------------------------------------------------------------
// 4. Layer2: h2[P][128] = conv2( relu(bn1(h1)) ) via v_wmma_f32_16x16x32_f16.
//    One wave = 16 positions x 128 out channels (8 N-tiles, 2 K-tiles).
// ---------------------------------------------------------------------------
__global__ __launch_bounds__(256) void layer2_kernel(
    const _Float16* __restrict__ h1, const float* __restrict__ W2,
    const float* __restrict__ b2, const float* __restrict__ a1,
    const float* __restrict__ c1, _Float16* __restrict__ h2) {
  __shared__ _Float16 sW[C2 * C1];  // 16 KB
  __shared__ float sa[C1], sc[C1], sb2[C2];
  int tid = threadIdx.x;
  for (int i = tid; i < C2 * C1; i += 256) sW[i] = (_Float16)W2[i];
  if (tid < C1) { sa[tid] = a1[tid]; sc[tid] = c1[tid]; }
  if (tid < C2) sb2[tid] = b2[tid];
  __syncthreads();

  int wave = tid >> 5, lane = tid & 31;
  int m  = lane & 15;
  int hi = lane >> 4;
  long p0   = (long)blockIdx.x * 128 + wave * 16;
  long prow = p0 + m;

  // A fragments: 16x32 f16; lane<16: K {0..7,16..23}, lane>=16: K {8..15,24..31}
  v16h A[2];
#pragma unroll
  for (int kt = 0; kt < 2; kt++) {
    int cb = kt * 32 + hi * 8;
    const h8* src = (const h8*)(h1 + prow * C1 + cb);
    h8 lo  = src[0];  // cols cb..cb+7
    h8 hi8 = src[2];  // cols cb+16..cb+23
    v16h t;
#pragma unroll
    for (int e = 0; e < 8; e++) {
      int ca = cb + e, cbn = cb + 16 + e;
      float v = fmaxf(0.f, sa[ca] * (float)lo[e] + sc[ca]);
      float w = fmaxf(0.f, sa[cbn] * (float)hi8[e] + sc[cbn]);
      t[e]     = (_Float16)v;
      t[8 + e] = (_Float16)w;
    }
    A[kt] = t;
  }

#pragma unroll
  for (int t8 = 0; t8 < 8; t8++) {
    int ch = t8 * 16 + m;  // B col N = lane&15
    v8f acc;
    float bb = sb2[ch];
#pragma unroll
    for (int e = 0; e < 8; e++) acc[e] = bb;
#pragma unroll
    for (int kt = 0; kt < 2; kt++) {
      // B fragment: lane<16: K = kt*32 + {0..7,16..23} ; lane>=16: K = kt*32 + {8..15,24..31}
      const h8* wsrc = (const h8*)(sW + ch * C1 + kt * 32 + hi * 8);
      h8 lo = wsrc[0], hh = wsrc[2];
      v16h Bf;
#pragma unroll
      for (int e = 0; e < 8; e++) { Bf[e] = lo[e]; Bf[8 + e] = hh[e]; }
      acc = __builtin_amdgcn_wmma_f32_16x16x32_f16(false, A[kt], false, Bf,
                                                   (short)0, acc, false, false);
    }
    long rowbase = p0 + (hi ? 8 : 0);
#pragma unroll
    for (int j = 0; j < 8; j++)
      h2[(rowbase + j) * C2 + ch] = (_Float16)acc[j];
  }
}

// ---------------------------------------------------------------------------
// 6. Layer3: h3[P][128] = conv3( relu(bn2(h2)) ), K=128 (4 K-tiles).
// ---------------------------------------------------------------------------
__global__ __launch_bounds__(256) void layer3_kernel(
    const _Float16* __restrict__ h2, const float* __restrict__ W3,
    const float* __restrict__ b3, const float* __restrict__ a2,
    const float* __restrict__ c2, _Float16* __restrict__ h3) {
  __shared__ _Float16 sW[C3 * C2];  // 32 KB
  __shared__ float sa[C2], sc[C2], sb3[C3];
  int tid = threadIdx.x;
  for (int i = tid; i < C3 * C2; i += 256) sW[i] = (_Float16)W3[i];
  if (tid < C2) { sa[tid] = a2[tid]; sc[tid] = c2[tid]; }
  if (tid < C3) sb3[tid] = b3[tid];
  __syncthreads();

  int wave = tid >> 5, lane = tid & 31;
  int m  = lane & 15;
  int hi = lane >> 4;
  long p0   = (long)blockIdx.x * 128 + wave * 16;
  long prow = p0 + m;

  v16h A[4];
#pragma unroll
  for (int kt = 0; kt < 4; kt++) {
    int cb = kt * 32 + hi * 8;
    const h8* src = (const h8*)(h2 + prow * C2 + cb);
    h8 lo  = src[0];
    h8 hi8 = src[2];
    v16h t;
#pragma unroll
    for (int e = 0; e < 8; e++) {
      int ca = cb + e, cbn = cb + 16 + e;
      float v = fmaxf(0.f, sa[ca] * (float)lo[e] + sc[ca]);
      float w = fmaxf(0.f, sa[cbn] * (float)hi8[e] + sc[cbn]);
      t[e]     = (_Float16)v;
      t[8 + e] = (_Float16)w;
    }
    A[kt] = t;
  }

#pragma unroll
  for (int t8 = 0; t8 < 8; t8++) {
    int ch = t8 * 16 + m;
    v8f acc;
    float bb = sb3[ch];
#pragma unroll
    for (int e = 0; e < 8; e++) acc[e] = bb;
#pragma unroll
    for (int kt = 0; kt < 4; kt++) {
      const h8* wsrc = (const h8*)(sW + ch * C2 + kt * 32 + hi * 8);
      h8 lo = wsrc[0], hh = wsrc[2];
      v16h Bf;
#pragma unroll
      for (int e = 0; e < 8; e++) { Bf[e] = lo[e]; Bf[8 + e] = hh[e]; }
      acc = __builtin_amdgcn_wmma_f32_16x16x32_f16(false, A[kt], false, Bf,
                                                   (short)0, acc, false, false);
    }
    long rowbase = p0 + (hi ? 8 : 0);
#pragma unroll
    for (int j = 0; j < 8; j++)
      h3[(rowbase + j) * C3 + ch] = (_Float16)acc[j];
  }
}

// ---------------------------------------------------------------------------
// 8. BN3 + max over k, LDS transpose for coalesced (B,128,N) output.
// ---------------------------------------------------------------------------
__global__ __launch_bounds__(256) void maxpool_kernel(
    const _Float16* __restrict__ h3, const float* __restrict__ a3,
    const float* __restrict__ c3, float* __restrict__ out) {
  __shared__ float tile[32][129];  // padded vs bank conflicts
  int b  = blockIdx.x >> 7;              // N/32 = 128 chunks per batch
  int n0 = (blockIdx.x & 127) * 32;
  int tid  = threadIdx.x;
  int o    = tid & 127;
  int half = tid >> 7;
  float a = a3[o], c = c3[o];
  for (int nn = half; nn < 32; nn += 2) {
    int n = n0 + nn;
    const _Float16* src = h3 + ((long)(b * NPTS + n) * KNN) * C3 + o;
    float mx = -__builtin_inff();
#pragma unroll
    for (int k = 0; k < KNN; k++) {
      float v = a * (float)src[k * C3] + c;
      mx = fmaxf(mx, v);
    }
    tile[nn][o] = mx;
  }
  __syncthreads();
  for (int r = 0; r < 16; r++) {
    int i = r * 256 + tid;
    int oo = i >> 5, j = i & 31;
    out[((long)(b * C3 + oo)) * NPTS + n0 + j] = tile[j][oo];
  }
}

// ---------------------------------------------------------------------------
// Host launcher. ws layout (bytes):
//   idx      : 0                       (P * 4      = 1,310,720)
//   h1  f16  : 1,310,720              (P * 64 * 2 = 41,943,040)
//   h2  f16  : 43,253,760             (P * 128* 2 = 83,886,080)
//   h3  f16  : 127,139,840            (P * 128* 2 = 83,886,080)
//   partials : 211,025,920            (512 * 256 * 4 = 524,288)
//   bn prm   : 211,550,208            (640 * 4)
// Total ~211.6 MB.
// ---------------------------------------------------------------------------
extern "C" void kernel_launch(void* const* d_in, const int* in_sizes, int n_in,
                              void* d_out, int out_size, void* d_ws, size_t ws_size,
                              hipStream_t stream) {
  const float* x   = (const float*)d_in[0];
  const float* W1  = (const float*)d_in[1];
  const float* b1  = (const float*)d_in[2];
  const float* g1  = (const float*)d_in[3];
  const float* be1 = (const float*)d_in[4];
  const float* W2  = (const float*)d_in[5];
  const float* b2  = (const float*)d_in[6];
  const float* g2  = (const float*)d_in[7];
  const float* be2 = (const float*)d_in[8];
  const float* W3  = (const float*)d_in[9];
  const float* b3  = (const float*)d_in[10];
  const float* g3  = (const float*)d_in[11];
  const float* be3 = (const float*)d_in[12];
  float* out = (float*)d_out;

  char* ws = (char*)d_ws;
  int*      idx      = (int*)ws;
  _Float16* h1       = (_Float16*)(ws + 1310720L);
  _Float16* h2       = (_Float16*)(ws + 43253760L);
  _Float16* h3       = (_Float16*)(ws + 127139840L);
  float*    partials = (float*)(ws + 211025920L);
  float*    prm      = (float*)(ws + 211550208L);
  float *a1 = prm, *c1 = prm + 64, *a2 = prm + 128, *c2 = prm + 256,
        *a3 = prm + 384, *c3 = prm + 512;

  // 1. KNN indices
  knn_kernel<<<BB * (NPTS / QPB), 256, 0, stream>>>(x, idx);
  // 2. features + conv1 -> h1
  feat_conv1_kernel<<<PPOS / 256, 256, 0, stream>>>(x, idx, W1, b1, h1);
  // 3. BN1 stats
  stats_partial_kernel<C1><<<STATBLKS, 256, 0, stream>>>(h1, partials);
  stats_final_kernel<<<1, C1, 0, stream>>>(partials, C1, g1, be1, a1, c1);
  // 4. layer2 (WMMA) -> h2
  layer2_kernel<<<PPOS / 128, 256, 0, stream>>>(h1, W2, b2, a1, c1, h2);
  // 5. BN2 stats
  stats_partial_kernel<C2><<<STATBLKS, 256, 0, stream>>>(h2, partials);
  stats_final_kernel<<<1, C2, 0, stream>>>(partials, C2, g2, be2, a2, c2);
  // 6. layer3 (WMMA) -> h3
  layer3_kernel<<<PPOS / 128, 256, 0, stream>>>(h2, W3, b3, a2, c2, h3);
  // 7. BN3 stats
  stats_partial_kernel<C3><<<STATBLKS, 256, 0, stream>>>(h3, partials);
  stats_final_kernel<<<1, C3, 0, stream>>>(partials, C3, g3, be3, a3, c3);
  // 8. BN3 + max over k -> out
  maxpool_kernel<<<BB * (NPTS / 32), 256, 0, stream>>>(h3, a3, c3, out);
}
